// ExtractorLoss_62148176773527
// MI455X (gfx1250) — matrix operand order repaired and checked by reading, not compiled, not run on hardware
//
#include <hip/hip_runtime.h>
#include <hip/hip_bf16.h>

// CDNA5 / gfx1250 wave32 WMMA implementation of ExtractorLoss.
//
// psd(b,g) = (x[b,:] . cos_g)^2 + (x[b,:] . sin_g)^2  -> GEMM [B,T]x[T,2*Ng]
// done with v_wmma_f32_16x16x32_f16 (f16 inputs, f32 accumulation).
//
// Padded geometry (compile-time): columns 0..223 = cos (201 real + pad),
// columns 224..447 = sin. N tiles = 28. T padded up to multiple of 32.
// Per-wave register tile: 32(M) x 64(N) -> 8 accumulators, 8 WMMA per
// 6 fragment loads per K step (1.5 b128 loads / WMMA).

typedef __attribute__((ext_vector_type(16))) _Float16 v16h;
typedef __attribute__((ext_vector_type(8)))  float    v8f;

#define NHALF        224      // padded columns per (cos|sin) block (14 tiles)
#define NCOLS        448      // total padded output columns
#define NTILES_TOTAL 28       // 448 / 16

union PackH2 { _Float16 h[2]; unsigned u; };

// ---------------------------------------------------------------------------
// Kernel 1: build DFT basis directly in WMMA B-fragment layout (f16 packed).
// B (32x16) packing used here, contiguous K (same as A): lanes 0-15 hold K=0..15 (2 per VGPR),
// lanes 16-31 hold K=16..31; column = ntile*16 + (lane&15).
// blob layout: [kt][ntile][lane][vgpr] as u32 (two packed f16).
// ---------------------------------------------------------------------------
__global__ void __launch_bounds__(256)
build_basis_kernel(unsigned* __restrict__ bfrag,
                   const int* __restrict__ fs_p,
                   const int* __restrict__ samp_p,
                   const int* __restrict__ fmin_p,
                   const int* __restrict__ fmax_p,
                   int T, int ktiles)
{
    int tid   = blockIdx.x * blockDim.x + threadIdx.x;
    int total = ktiles * NTILES_TOTAL * 256;
    if (tid >= total) return;

    int v    = tid & 7;
    int lane = (tid >> 3) & 31;
    int nt   = (tid >> 8) % NTILES_TOTAL;
    int kt   = tid / (NTILES_TOTAL * 256);

    const int   fs     = fs_p[0];
    const int   samp   = samp_p[0];
    const int   fmin   = fmin_p[0];
    const int   fmax   = fmax_p[0];
    const int   n_grid = (fmax - fmin) / samp + 1;       // 201
    const int   modP   = 60 * fs;                        // exact phase period
    const float ph_sc  = 6.28318530717958647692f / (float)modP;

    int c  = nt * 16 + (lane & 15);                      // combined column
    int kl = ((lane < 16) ? 0 : 16) + 2 * v;             // K within tile
    int t0 = kt * 32 + kl;

    int g, is_sin;
    if (c < NHALF) { g = c;         is_sin = 0; }
    else           { g = c - NHALF; is_sin = 1; }

    PackH2 p;
#pragma unroll
    for (int j = 0; j < 2; ++j) {
        int   t   = t0 + j;
        float val = 0.0f;
        if (g < n_grid && t < T) {
            int   fbpm = fmin + samp * g;
            int   m    = (fbpm * t) % modP;              // exact reduction
            float ph   = ph_sc * (float)m;
            val = is_sin ? sinf(ph) : cosf(ph);
        }
        p.h[j] = (_Float16)val;
    }
    bfrag[tid] = p.u;
}

// ---------------------------------------------------------------------------
// Kernel 2: pack x (f32 row-major) into WMMA A-fragment layout (f16 packed).
// A (16x32) packing used here, contiguous K (same as B): row M = lane&15;
// lanes 0-15: VGPR0-7 -> K=0..15 (2 per VGPR);
// lanes 16-31: VGPR0-7 -> K=16..31.
// blob layout: [mtile][kt][lane][vgpr] as u32.
// ---------------------------------------------------------------------------
__global__ void __launch_bounds__(256)
pack_a_kernel(unsigned* __restrict__ afrag,
              const float* __restrict__ x,
              int T, int ktiles, int mtiles)
{
    int tid   = blockIdx.x * blockDim.x + threadIdx.x;
    int total = mtiles * ktiles * 256;
    if (tid >= total) return;

    int v     = tid & 7;
    int lane  = (tid >> 3) & 31;
    int kt    = (tid >> 8) % ktiles;
    int mtile = tid / (ktiles * 256);

    int lo  = (lane < 16) ? 0 : 16;
    int kl  = lo + 2 * v;
    int row = mtile * 16 + (lane & 15);
    int t0  = kt * 32 + kl;

    const float* xr = x + (size_t)row * T;
    PackH2 p;
    p.h[0] = (_Float16)((t0     < T) ? xr[t0]     : 0.0f);
    p.h[1] = (_Float16)((t0 + 1 < T) ? xr[t0 + 1] : 0.0f);
    afrag[tid] = p.u;
}

// ---------------------------------------------------------------------------
// Kernel 3: the GEMM. 8 waves / block; wave w owns M-tiles 2*(blockIdx.x*8+w)
// and +1, plus four N-tiles (blockIdx.y*4 .. +3): a 32x64 register tile,
// 8 x v_wmma_f32_16x16x32_f16 per K step for 6 fragment loads.
// All 8 waves share the same 4 B fragments -> L1/L2 reuse; x fits in L2.
// ---------------------------------------------------------------------------
__global__ void __launch_bounds__(256)
wmma_gemm_kernel(const unsigned* __restrict__ afrag,
                 const unsigned* __restrict__ bfrag,
                 float* __restrict__ C, int ktiles)
{
    int lane = threadIdx.x & 31;
    int wave = threadIdx.x >> 5;
    int mt0  = (blockIdx.x * 8 + wave) * 2;   // two consecutive M tiles
    int nt0  = blockIdx.y * 4;                // four consecutive N tiles

    const unsigned* ap0 = afrag + (((size_t)mt0       * ktiles) * 32 + lane) * 8;
    const unsigned* ap1 = afrag + (((size_t)(mt0 + 1) * ktiles) * 32 + lane) * 8;
    const unsigned* bp  = bfrag + (((size_t)nt0 * 32) + lane) * 8;
    const size_t a_stride = 256;                          // dwords per K tile
    const size_t b_stride = (size_t)NTILES_TOTAL * 256;

    v8f acc[2][4] = {};
    for (int kt = 0; kt < ktiles; ++kt) {
        v16h a0 = *reinterpret_cast<const v16h*>(ap0);
        v16h a1 = *reinterpret_cast<const v16h*>(ap1);
        v16h b0 = *reinterpret_cast<const v16h*>(bp);
        v16h b1 = *reinterpret_cast<const v16h*>(bp + 256);
        v16h b2 = *reinterpret_cast<const v16h*>(bp + 512);
        v16h b3 = *reinterpret_cast<const v16h*>(bp + 768);

        acc[0][0] = __builtin_amdgcn_wmma_f32_16x16x32_f16(
                        false, a0, false, b0, (short)0, acc[0][0], false, false);
        acc[0][1] = __builtin_amdgcn_wmma_f32_16x16x32_f16(
                        false, a0, false, b1, (short)0, acc[0][1], false, false);
        acc[0][2] = __builtin_amdgcn_wmma_f32_16x16x32_f16(
                        false, a0, false, b2, (short)0, acc[0][2], false, false);
        acc[0][3] = __builtin_amdgcn_wmma_f32_16x16x32_f16(
                        false, a0, false, b3, (short)0, acc[0][3], false, false);
        acc[1][0] = __builtin_amdgcn_wmma_f32_16x16x32_f16(
                        false, a1, false, b0, (short)0, acc[1][0], false, false);
        acc[1][1] = __builtin_amdgcn_wmma_f32_16x16x32_f16(
                        false, a1, false, b1, (short)0, acc[1][1], false, false);
        acc[1][2] = __builtin_amdgcn_wmma_f32_16x16x32_f16(
                        false, a1, false, b2, (short)0, acc[1][2], false, false);
        acc[1][3] = __builtin_amdgcn_wmma_f32_16x16x32_f16(
                        false, a1, false, b3, (short)0, acc[1][3], false, false);

        ap0 += a_stride;
        ap1 += a_stride;
        bp  += b_stride;
    }

    // C/D fragment layout: N = lane&15; M = r + 8*(lane>>4) for VGPR r.
    int n = lane & 15, half = lane >> 4;
#pragma unroll
    for (int m = 0; m < 2; ++m) {
        float* crow = C + (size_t)((mt0 + m) * 16) * NCOLS;
#pragma unroll
        for (int j = 0; j < 4; ++j) {
#pragma unroll
            for (int r = 0; r < 8; ++r) {
                crow[(size_t)(r + 8 * half) * NCOLS + (nt0 + j) * 16 + n] =
                    acc[m][j][r];
            }
        }
    }
}

// ---------------------------------------------------------------------------
// Kernel 4: per-row SNR. One wave32 per batch row; shfl_xor reduction.
// ---------------------------------------------------------------------------
__global__ void __launch_bounds__(256)
row_snr_kernel(const float* __restrict__ C,
               const int* __restrict__ ftrue,
               const int* __restrict__ delta_p,
               const int* __restrict__ samp_p,
               const int* __restrict__ fmin_p,
               const int* __restrict__ fmax_p,
               float* __restrict__ snr)
{
    int lane = threadIdx.x & 31;
    int wave = threadIdx.x >> 5;
    int row  = blockIdx.x * 8 + wave;

    const int delta  = delta_p[0];
    const int samp   = samp_p[0];
    const int fmin   = fmin_p[0];
    const int fmax   = fmax_p[0];
    const int n_grid = (fmax - fmin) / samp + 1;          // 201
    const int n_w    = 2 * delta / samp + 1;              // 11
    const int ft     = ftrue[row];

    const float* cr = C + (size_t)row * NCOLS;
    float sw = 0.0f, su = 0.0f;
    for (int g = lane; g < n_grid; g += 32) {
        float a = cr[g];
        float b = cr[NHALF + g];
        float p = a * a + b * b;
        int d = fmin + samp * g - ft;
        if (d < 0) d = -d;
        if (d <= delta) sw += p; else su += p;
    }
#pragma unroll
    for (int off = 16; off > 0; off >>= 1) {
        sw += __shfl_xor(sw, off, 32);
        su += __shfl_xor(su, off, 32);
    }
    if (lane == 0) {
        float t1 = sw / (float)n_w;
        float t2 = su / (float)(n_grid - n_w);
        snr[row] = 10.0f * log10f(t1 / t2);
    }
}

// ---------------------------------------------------------------------------
// Kernel 5: final mean -> out[0] = -mean(snr)
// ---------------------------------------------------------------------------
__global__ void __launch_bounds__(256)
final_mean_kernel(const float* __restrict__ snr, float* __restrict__ out, int B)
{
    __shared__ float sm[256];
    float s = 0.0f;
    for (int i = threadIdx.x; i < B; i += 256) s += snr[i];
    sm[threadIdx.x] = s;
    __syncthreads();
    for (int st = 128; st > 0; st >>= 1) {
        if ((int)threadIdx.x < st) sm[threadIdx.x] += sm[threadIdx.x + st];
        __syncthreads();
    }
    if (threadIdx.x == 0) out[0] = -sm[0] / (float)B;
}

// ---------------------------------------------------------------------------
extern "C" void kernel_launch(void* const* d_in, const int* in_sizes, int n_in,
                              void* d_out, int out_size, void* d_ws, size_t ws_size,
                              hipStream_t stream)
{
    const float* x       = (const float*)d_in[0];
    const int*   ftrue   = (const int*)d_in[1];
    const int*   fs_p    = (const int*)d_in[2];
    const int*   delta_p = (const int*)d_in[3];
    const int*   samp_p  = (const int*)d_in[4];
    const int*   fmin_p  = (const int*)d_in[5];
    const int*   fmax_p  = (const int*)d_in[6];
    float*       out     = (float*)d_out;

    const int B      = in_sizes[1];          // 4096
    const int T      = in_sizes[0] / B;      // 3600
    const int ktiles = (T + 31) / 32;        // 113
    const int mtiles = B / 16;               // 256

    // Workspace carve-up (256B aligned blocks).
    char*  ws       = (char*)d_ws;
    size_t a_dwords = (size_t)mtiles * ktiles * 256;            // A frag blob
    size_t b_dwords = (size_t)ktiles * NTILES_TOTAL * 256;      // B frag blob
    unsigned* afrag = (unsigned*)ws;
    unsigned* bfrag = (unsigned*)(ws + ((a_dwords * 4 + 255) & ~(size_t)255));
    float*    C     = (float*)((char*)bfrag + ((b_dwords * 4 + 255) & ~(size_t)255));
    float*    snr   = (float*)((char*)C + (((size_t)B * NCOLS * 4 + 255) & ~(size_t)255));

    int bblocks = (int)((b_dwords + 255) / 256);
    build_basis_kernel<<<bblocks, 256, 0, stream>>>(bfrag, fs_p, samp_p, fmin_p,
                                                    fmax_p, T, ktiles);

    int ablocks = (int)((a_dwords + 255) / 256);
    pack_a_kernel<<<ablocks, 256, 0, stream>>>(afrag, x, T, ktiles, mtiles);

    dim3 ggrid(mtiles / 16, NTILES_TOTAL / 4);   // (16, 7)
    wmma_gemm_kernel<<<ggrid, 256, 0, stream>>>(afrag, bfrag, C, ktiles);

    row_snr_kernel<<<B / 8, 256, 0, stream>>>(C, ftrue, delta_p, samp_p,
                                              fmin_p, fmax_p, snr);

    final_mean_kernel<<<1, 256, 0, stream>>>(snr, out, B);
}
